// ByteMultiHeadSelfAttention_29618094474208
// MI455X (gfx1250) — matrix operand order repaired and checked by reading, not compiled, not run on hardware
//
#include <hip/hip_runtime.h>

#define DI __device__ __forceinline__

typedef __bf16 bf16;
typedef __attribute__((ext_vector_type(16))) __bf16 v16b;
typedef __attribute__((ext_vector_type(8)))  __bf16 v8b;
typedef __attribute__((ext_vector_type(8)))  float  v8f;

constexpr int B_ = 2, S_ = 2048, E_ = 2048, H_ = 16, KV_ = 4, D_ = 128, WIN_ = 512;
constexpr int MS  = B_ * S_;    // 4096 tokens
constexpr int KVD = KV_ * D_;   // 512
constexpr float SCALE = 0.08838834764831845f;  // 1/sqrt(128)
constexpr float NEGV  = -1e9f;

// ---- helpers ---------------------------------------------------------------

DI bf16 f2bf(float f) {                      // round-to-nearest-even f32 -> bf16
  unsigned u = __builtin_bit_cast(unsigned, f);
  u += 0x7fffu + ((u >> 16) & 1u);
  unsigned short h = (unsigned short)(u >> 16);
  return __builtin_bit_cast(bf16, h);
}

// A-fragment (16x32 bf16): lane holds row M=l16; K = (i<8?i:i+8)+8*half
// => two contiguous 8-elem (16B) chunks at offsets 8*half and 8*half+16.
DI v16b ldfragA(const bf16* rowp, int off) {
  v8b lo = *(const v8b*)(rowp + off);
  v8b hi = *(const v8b*)(rowp + off + 16);
  return __builtin_shufflevector(lo, hi, 0,1,2,3,4,5,6,7,8,9,10,11,12,13,14,15);
}

DI v8f wmma_bf16(v16b a, v16b b, v8f c) {
  return __builtin_amdgcn_wmma_f32_16x16x32_bf16(false, a, false, b, (short)0, c,
                                                 false, false);
}

// ---- elementwise / layout kernels ------------------------------------------

__global__ void k_cast_bf16(const float* __restrict__ in, bf16* __restrict__ out,
                            int n) {
  int i = blockIdx.x * blockDim.x + threadIdx.x;
  if (i < n) out[i] = f2bf(in[i]);
}

// in: [K][N] f32 (row-major)  ->  out: [N][K] bf16 (row-major)
__global__ void k_transpose_cast(const float* __restrict__ in,
                                 bf16* __restrict__ out, int K, int N) {
  __shared__ float tile[32][33];
  int tx = threadIdx.x & 31, ty = threadIdx.x >> 5;
  int n0 = blockIdx.x * 32, k0 = blockIdx.y * 32;
  for (int r = ty; r < 32; r += 8)
    tile[r][tx] = in[(size_t)(k0 + r) * N + (n0 + tx)];
  __syncthreads();
  for (int r = ty; r < 32; r += 8)
    out[(size_t)(n0 + r) * K + (k0 + tx)] = f2bf(tile[tx][r]);
}

// RoPE on Q and re-layout [B,S,H,D] f32 -> [B,H,S,D] bf16
__global__ void k_rope_q(const float* __restrict__ Qf, bf16* __restrict__ Qb) {
  int idx = blockIdx.x * blockDim.x + threadIdx.x;  // B*S*H*64
  int m = idx & 63;
  int h = (idx >> 6) & (H_ - 1);
  int s = (idx >> 10) & (S_ - 1);
  int b = idx >> 21;
  float ang = (float)s * __expf(-(float)m * (9.2103403719761836f / 64.f));
  float sn, cs; __sincosf(ang, &sn, &cs);
  const float* src = Qf + ((size_t)(b * S_ + s) * H_ + h) * D_;
  float t1 = src[m], t2 = src[m + 64];
  bf16* dst = Qb + ((size_t)(b * H_ + h) * S_ + s) * D_;
  dst[m]      = f2bf(t1 * cs - t2 * sn);
  dst[m + 64] = f2bf(t1 * sn + t2 * cs);
}

// RoPE on K and re-layout [B,S,KV,D] f32 -> [B,KV,S,D] bf16
__global__ void k_rope_k(const float* __restrict__ Kf, bf16* __restrict__ Kb) {
  int idx = blockIdx.x * blockDim.x + threadIdx.x;  // B*S*KV*64
  int m = idx & 63;
  int g = (idx >> 6) & (KV_ - 1);
  int s = (idx >> 8) & (S_ - 1);
  int b = idx >> 19;
  float ang = (float)s * __expf(-(float)m * (9.2103403719761836f / 64.f));
  float sn, cs; __sincosf(ang, &sn, &cs);
  const float* src = Kf + ((size_t)(b * S_ + s) * KV_ + g) * D_;
  float t1 = src[m], t2 = src[m + 64];
  bf16* dst = Kb + ((size_t)(b * KV_ + g) * S_ + s) * D_;
  dst[m]      = f2bf(t1 * cs - t2 * sn);
  dst[m + 64] = f2bf(t1 * sn + t2 * cs);
}

// V re-layout [B,S,KV,D] f32 -> transposed [B,KV,D,S] bf16
__global__ void k_vtrans(const float* __restrict__ Vf, bf16* __restrict__ Vt) {
  int idx = blockIdx.x * blockDim.x + threadIdx.x;  // B*S*KV*D
  int d = idx & 127;
  int g = (idx >> 7) & (KV_ - 1);
  int s = (idx >> 9) & (S_ - 1);
  int b = idx >> 20;
  Vt[((size_t)(b * KV_ + g) * D_ + d) * S_ + s] = f2bf(Vf[idx]);
}

// ---- generic WMMA GEMM: C[M,N] f32 = A[M,K] bf16 * Bt[N,K]^T bf16 ----------
// block = 256 threads = 8 waves; block tile 128x128; wave tile 32x64.

__global__ __launch_bounds__(256)
void k_gemm_bf16(const bf16* __restrict__ A, const bf16* __restrict__ Bt,
                 float* __restrict__ C, int M, int N, int K) {
  int lane = threadIdx.x & 31, w = threadIdx.x >> 5;
  int l16 = lane & 15, half = lane >> 4;
  int wm = blockIdx.y * 128 + (w & 3) * 32;
  int wn = blockIdx.x * 128 + (w >> 2) * 64;

  const bf16* a0 = A + (size_t)(wm + l16) * K;
  const bf16* a1 = a0 + (size_t)16 * K;
  const bf16* b0 = Bt + (size_t)(wn + l16) * K;

  v8f acc[2][4] = {};
  for (int k = 0; k < K; k += 32) {
    int offA = k + half * 8;
    int offB = k + half * 8;
    v16b af0 = ldfragA(a0, offA);
    v16b af1 = ldfragA(a1, offA);
#pragma unroll
    for (int j = 0; j < 4; ++j) {
      v16b bf = ldfragA(b0 + (size_t)(j * 16) * K, offB);
      acc[0][j] = wmma_bf16(af0, bf, acc[0][j]);
      acc[1][j] = wmma_bf16(af1, bf, acc[1][j]);
    }
  }
#pragma unroll
  for (int a = 0; a < 2; ++a)
#pragma unroll
    for (int j = 0; j < 4; ++j)
#pragma unroll
      for (int v = 0; v < 8; ++v)
        C[(size_t)(wm + a * 16 + v + 8 * half) * N + wn + j * 16 + l16] =
            acc[a][j][v];
}

// ---- flash attention with sliding-window causal mask -----------------------
// grid (S/64, H, B), block 128 = 4 waves; wave handles 16 query rows.

__global__ __launch_bounds__(128)
void k_attn(const bf16* __restrict__ Qb, const bf16* __restrict__ Kb,
            const bf16* __restrict__ Vt, bf16* __restrict__ Ob) {
  __shared__ bf16 plds[4][16 * 32];  // per-wave P tile (16 q x 32 k)
  int lane = threadIdx.x & 31, w = threadIdx.x >> 5;
  int l16 = lane & 15, half = lane >> 4;
  int h = blockIdx.y, b = blockIdx.z, g = h >> 2;
  int q0 = blockIdx.x * 64 + w * 16;

  const bf16* qrow = Qb + ((size_t)(b * H_ + h) * S_ + q0 + l16) * D_;
  v16b qf[4];
#pragma unroll
  for (int c = 0; c < 4; ++c) qf[c] = ldfragA(qrow, c * 32 + half * 8);

  const bf16* kbase = Kb + (size_t)(b * KV_ + g) * S_ * D_;
  const bf16* vbase = Vt + (size_t)(b * KV_ + g) * D_ * S_;

  v8f acc[8] = {};
  float rmax[8], rsum[8];
#pragma unroll
  for (int v = 0; v < 8; ++v) { rmax[v] = -1e30f; rsum[v] = 0.f; }

  int kstart = q0 - WIN_ + 1; if (kstart < 0) kstart = 0; kstart &= ~31;
  int kend = q0 + 15;

  for (int kt = kstart; kt <= kend; kt += 32) {
    // S = Q * K^T for two 16-key tiles
    v8f s0 = {}, s1 = {};
    const bf16* krow0 = kbase + (size_t)(kt + l16) * D_;
    const bf16* krow1 = krow0 + (size_t)16 * D_;
#pragma unroll
    for (int c = 0; c < 4; ++c) {
      int off = c * 32 + half * 8;
      v16b kf0 = ldfragA(krow0, off);
      v16b kf1 = ldfragA(krow1, off);
      s0 = wmma_bf16(qf[c], kf0, s0);
      s1 = wmma_bf16(qf[c], kf1, s1);
    }
    // mask + online softmax (row m = v + 8*half lives in one 16-lane half)
#pragma unroll
    for (int v = 0; v < 8; ++v) {
      int i  = q0 + v + 8 * half;
      int j0 = kt + l16, j1 = j0 + 16;
      float e0 = s0[v] * SCALE + (((j0 <= i) && (i - j0 < WIN_)) ? 0.f : NEGV);
      float e1 = s1[v] * SCALE + (((j1 <= i) && (i - j1 < WIN_)) ? 0.f : NEGV);
      float mx = fmaxf(e0, e1);
#pragma unroll
      for (int o = 8; o >= 1; o >>= 1) mx = fmaxf(mx, __shfl_xor(mx, o));
      float nm   = fmaxf(rmax[v], mx);
      float corr = __expf(rmax[v] - nm);
      rmax[v] = nm;
      float p0 = __expf(e0 - nm);
      float p1 = __expf(e1 - nm);
      float ps = p0 + p1;
#pragma unroll
      for (int o = 8; o >= 1; o >>= 1) ps += __shfl_xor(ps, o);
      rsum[v] = rsum[v] * corr + ps;
#pragma unroll
      for (int t = 0; t < 8; ++t) acc[t][v] *= corr;
      int r = v + 8 * half;
      plds[w][r * 32 + l16]      = f2bf(p0);
      plds[w][r * 32 + 16 + l16] = f2bf(p1);
    }
    // same-wave DS ops are in-order; wait stores then reload in A-frag layout
    asm volatile("s_wait_dscnt 0x0" ::: "memory");
    v8b plo = *(const v8b*)(&plds[w][l16 * 32 + half * 8]);
    v8b phi = *(const v8b*)(&plds[w][l16 * 32 + half * 8 + 16]);
    v16b pf = __builtin_shufflevector(plo, phi,
                                      0,1,2,3,4,5,6,7,8,9,10,11,12,13,14,15);
    // O += P * V
#pragma unroll
    for (int t = 0; t < 8; ++t) {
      const bf16* vrow = vbase + (size_t)(t * 16 + l16) * S_;
      v16b vf = ldfragA(vrow, kt + half * 8);
      acc[t] = wmma_bf16(pf, vf, acc[t]);
    }
  }
  // normalize and emit bf16 rows into [B*S, H*D]
#pragma unroll
  for (int t = 0; t < 8; ++t)
#pragma unroll
    for (int v = 0; v < 8; ++v) {
      int row = q0 + v + 8 * half;
      Ob[(size_t)(b * S_ + row) * E_ + h * D_ + t * 16 + l16] =
          f2bf(acc[t][v] / rsum[v]);
    }
}

// ---- launcher --------------------------------------------------------------

extern "C" void kernel_launch(void* const* d_in, const int* in_sizes, int n_in,
                              void* d_out, int out_size, void* d_ws,
                              size_t ws_size, hipStream_t stream) {
  (void)in_sizes; (void)n_in; (void)out_size; (void)ws_size;
  const float* x  = (const float*)d_in[0];
  const float* Wq = (const float*)d_in[1];
  const float* Wk = (const float*)d_in[2];
  const float* Wv = (const float*)d_in[3];
  const float* Wo = (const float*)d_in[4];
  float* out = (float*)d_out;

  char* base = (char*)d_ws;
  size_t off = 0;
  auto alloc = [&](size_t bytes) -> void* {
    void* r = base + off;
    off += (bytes + 255) & ~(size_t)255;
    return r;
  };
  bf16*  xb  = (bf16*) alloc((size_t)MS  * E_  * 2);
  bf16*  WqT = (bf16*) alloc((size_t)E_  * E_  * 2);
  bf16*  WkT = (bf16*) alloc((size_t)KVD * E_  * 2);
  bf16*  WvT = (bf16*) alloc((size_t)KVD * E_  * 2);
  bf16*  WoT = (bf16*) alloc((size_t)E_  * E_  * 2);
  float* Qf  = (float*)alloc((size_t)MS  * E_  * 4);
  float* Kf  = (float*)alloc((size_t)MS  * KVD * 4);
  float* Vf  = (float*)alloc((size_t)MS  * KVD * 4);
  bf16*  Qb  = (bf16*) alloc((size_t)MS  * E_  * 2);
  bf16*  Kb  = (bf16*) alloc((size_t)MS  * KVD * 2);
  bf16*  Vtb = (bf16*) alloc((size_t)MS  * KVD * 2 + 256);
  bf16*  Ob  = (bf16*)Qf;  // Qf fully consumed by RoPE before attention writes

  // 1) casts / weight transposes (bf16, W^T layout for B-fragments)
  {
    int n = MS * E_;
    k_cast_bf16<<<(n + 255) / 256, 256, 0, stream>>>(x, xb, n);
  }
  k_transpose_cast<<<dim3(E_ / 32,  E_ / 32), 256, 0, stream>>>(Wq, WqT, E_, E_);
  k_transpose_cast<<<dim3(KVD / 32, E_ / 32), 256, 0, stream>>>(Wk, WkT, E_, KVD);
  k_transpose_cast<<<dim3(KVD / 32, E_ / 32), 256, 0, stream>>>(Wv, WvT, E_, KVD);
  k_transpose_cast<<<dim3(E_ / 32,  E_ / 32), 256, 0, stream>>>(Wo, WoT, E_, E_);

  // 2) QKV projections (WMMA GEMMs)
  k_gemm_bf16<<<dim3(E_ / 128,  MS / 128), 256, 0, stream>>>(xb, WqT, Qf, MS, E_,  E_);
  k_gemm_bf16<<<dim3(KVD / 128, MS / 128), 256, 0, stream>>>(xb, WkT, Kf, MS, KVD, E_);
  k_gemm_bf16<<<dim3(KVD / 128, MS / 128), 256, 0, stream>>>(xb, WvT, Vf, MS, KVD, E_);

  // 3) RoPE + attention-friendly layouts
  k_rope_q<<<(B_ * S_ * H_  * 64) / 256, 256, 0, stream>>>(Qf, Qb);
  k_rope_k<<<(B_ * S_ * KV_ * 64) / 256, 256, 0, stream>>>(Kf, Kb);
  k_vtrans<<<(B_ * S_ * KV_ * D_) / 256, 256, 0, stream>>>(Vf, Vtb);

  // 4) sliding-window flash attention (WMMA)
  k_attn<<<dim3(S_ / 64, H_, B_), 128, 0, stream>>>(Qb, Kb, Vtb, Ob);

  // 5) output projection (WMMA GEMM, f32 result)
  k_gemm_bf16<<<dim3(E_ / 128, MS / 128), 256, 0, stream>>>(Ob, WoT, out, MS, E_, E_);
}
